// GroupedQuaryAttention_56831007261007
// MI455X (gfx1250) — matrix operand order repaired and checked by reading, not compiled, not run on hardware
//
#include <hip/hip_runtime.h>
#include <hip/hip_bf16.h>

// ---------------------------------------------------------------------------
// GroupedQueryAttention for MI455X (gfx1250), wave32 + WMMA f16->f32.
// B=2, S=2048, DIM=2048, HQ=32, HKV=8, HEAD_DIM=64, GROUP=4.
// ---------------------------------------------------------------------------

typedef _Float16 v16h __attribute__((ext_vector_type(16)));
typedef _Float16 v8h  __attribute__((ext_vector_type(8)));
typedef _Float16 v4h  __attribute__((ext_vector_type(4)));
typedef float    v8f  __attribute__((ext_vector_type(8)));
typedef float    v4f  __attribute__((ext_vector_type(4)));

#define GQA_B    2
#define GQA_S    2048
#define GQA_DIM  2048
#define GQA_HQ   32
#define GQA_HKV  8
#define GQA_HD   64
#define GQA_M    (GQA_B * GQA_S)      // 4096 tokens
#define GQA_KVD  (GQA_HKV * GQA_HD)   // 512

template <class T, class U> struct same_t { static constexpr bool value = false; };
template <class T>          struct same_t<T, T> { static constexpr bool value = true; };

__device__ __forceinline__ v16h cat16(v8h lo, v8h hi) {
    v16h r;
#pragma unroll
    for (int i = 0; i < 8; ++i) { r[i] = lo[i]; r[i + 8] = hi[i]; }
    return r;
}

__device__ __forceinline__ v8f wmma_f16(v16h a, v16h b, v8f c) {
    return __builtin_amdgcn_wmma_f32_16x16x32_f16(false, a, false, b,
                                                  (short)0, c, false, false);
}

// ---------------------------------------------------------------------------
// GEMM: C[M,N] = cast<OT>( A[M,K] * W[K,N] + bias[N] )
// Block 256 threads = 8 waves; tile 128x128; per-wave 64x32 (4x2 WMMA frags).
// Global->register staging is software-pipelined one K-step ahead so HBM
// latency hides behind WMMA; W is stored transposed in LDS so both A and B
// fragments are contiguous ds_load_b128 matching the ISA 16-bit layouts.
// ---------------------------------------------------------------------------
template <typename AT, typename OT>
__global__ __launch_bounds__(256) void gqa_gemm_wmma(
    const AT* __restrict__ A, const float* __restrict__ W,
    const float* __restrict__ bias, OT* __restrict__ C,
    int M, int N, int K)
{
    constexpr int BM = 128, BN = 128, BK = 32, PAD = 8;
    __shared__ _Float16 As[BM][BK + PAD];   // row-major (A layout)
    __shared__ _Float16 Wt[BN][BK + PAD];   // transposed W (n-major)

    const int tid  = threadIdx.x;
    const int l    = tid & 31;
    const int wid  = tid >> 5;
    const int wmv  = wid & 1;       // 2 waves along M (64 rows each)
    const int wnv  = wid >> 1;      // 4 waves along N (32 cols each)
    const int lm   = l & 15;
    const bool lo16 = (l < 16);
    const int m0 = blockIdx.y * BM;
    const int n0 = blockIdx.x * BN;

    v8f acc[4][2] = {};

    v4f aReg[4];   // f32-A staging (16 floats/thread)
    v8h aRegH[2];  // f16-A staging (16 halves/thread)
    v4f wReg[4];   // W staging (16 floats/thread)

    auto loadA = [&](int k0) {
        if constexpr (same_t<AT, float>::value) {
#pragma unroll
            for (int it = 0; it < 4; ++it) {
                int qq = tid + it * 256;           // 1024 float-quads
                int row = qq >> 3, c4 = (qq & 7) * 4;
                aReg[it] = *(const v4f*)&A[(size_t)(m0 + row) * K + k0 + c4];
            }
        } else {
#pragma unroll
            for (int it = 0; it < 2; ++it) {
                int qq = tid + it * 256;           // 512 half-octets
                int row = qq >> 2, c8 = (qq & 3) * 8;
                aRegH[it] = *(const v8h*)&A[(size_t)(m0 + row) * K + k0 + c8];
            }
        }
    };
    auto loadW = [&](int k0) {
#pragma unroll
        for (int it = 0; it < 4; ++it) {
            int qq = tid + it * 256;               // 1024 float-quads
            int kr = qq >> 5, nc = (qq & 31) * 4;
            wReg[it] = *(const v4f*)&W[(size_t)(k0 + kr) * N + n0 + nc];
        }
    };
    auto storeTiles = [&]() {
        if constexpr (same_t<AT, float>::value) {
#pragma unroll
            for (int it = 0; it < 4; ++it) {
                int qq = tid + it * 256;
                int row = qq >> 3, c4 = (qq & 7) * 4;
                v4h h = { (_Float16)aReg[it][0], (_Float16)aReg[it][1],
                          (_Float16)aReg[it][2], (_Float16)aReg[it][3] };
                *(v4h*)&As[row][c4] = h;
            }
        } else {
#pragma unroll
            for (int it = 0; it < 2; ++it) {
                int qq = tid + it * 256;
                int row = qq >> 2, c8 = (qq & 3) * 8;
                *(v8h*)&As[row][c8] = aRegH[it];
            }
        }
#pragma unroll
        for (int it = 0; it < 4; ++it) {
            int qq = tid + it * 256;
            int kr = qq >> 5, nc = (qq & 31) * 4;
#pragma unroll
            for (int j = 0; j < 4; ++j) Wt[nc + j][kr] = (_Float16)wReg[it][j];
        }
    };

    loadA(0);
    loadW(0);
    for (int k0 = 0; k0 < K; k0 += BK) {
        __syncthreads();                 // LDS reads of previous step complete
        storeTiles();
        __syncthreads();
        if (k0 + BK < K) {               // pipeline: fetch next tile now
            loadA(k0 + BK);
            loadW(k0 + BK);
            if (k0 + 2 * BK < K) {       // gfx1250 global_prefetch_b8, 2 ahead
                __builtin_prefetch(&A[(size_t)(m0 + (tid >> 1)) * K + k0 + 2 * BK], 0, 3);
                __builtin_prefetch(&W[(size_t)(k0 + 2 * BK + (tid >> 3)) * N + n0], 0, 3);
            }
        }

        const int koA = lo16 ? 0 : 8;
        v16h afr[4];
#pragma unroll
        for (int mi = 0; mi < 4; ++mi) {
            int m = wmv * 64 + mi * 16 + lm;
            afr[mi] = cat16(*(const v8h*)&As[m][koA],
                            *(const v8h*)&As[m][16 + koA]);
        }
        const int koB = lo16 ? 0 : 8;
#pragma unroll
        for (int ni = 0; ni < 2; ++ni) {
            int n = wnv * 32 + ni * 16 + lm;
            v16h bfr = cat16(*(const v8h*)&Wt[n][koB],
                             *(const v8h*)&Wt[n][koB + 16]);
#pragma unroll
            for (int mi = 0; mi < 4; ++mi)
                acc[mi][ni] = wmma_f16(afr[mi], bfr, acc[mi][ni]);
        }
    }

    // ---- epilogue: C/D layout -> global (+bias) ----
#pragma unroll
    for (int mi = 0; mi < 4; ++mi)
#pragma unroll
        for (int ni = 0; ni < 2; ++ni)
#pragma unroll
            for (int r = 0; r < 8; ++r) {
                int row = m0 + wmv * 64 + mi * 16 + (lo16 ? r : 8 + r);
                int col = n0 + wnv * 32 + ni * 16 + lm;
                float v = acc[mi][ni][r] + bias[col];
                C[(size_t)row * N + col] = (OT)v;
            }
}

// ---------------------------------------------------------------------------
// Flash attention: grid (S/128, HQ, B); block 256 = 8 waves; each wave owns
// 16 query rows with Q fragments resident in VGPRs. 64-key chunks (8 score
// WMMAs + 8 PV WMMAs per chunk) amortize the shuffle-based online softmax
// and the K/V tile staging, which is software-pipelined one chunk ahead.
// ---------------------------------------------------------------------------
__global__ __launch_bounds__(256) void gqa_flash_attn(
    const _Float16* __restrict__ Qb,   // [B*S, 2048] head-major
    const _Float16* __restrict__ Kb,   // [B*S, 512]
    const _Float16* __restrict__ Vb,   // [B*S, 512]
    _Float16* __restrict__ Ob)         // [B*S, 2048]
{
    __shared__ _Float16 Ks[64][72];        // [key][d]   (B layout for Q.K^T)
    __shared__ _Float16 Vt[64][72];        // [d][key]   (B layout for P.V)
    __shared__ _Float16 Ps[8][16][72];     // per-wave P staging (D -> A relayout)

    const int tid  = threadIdx.x;
    const int l    = tid & 31;
    const int wid  = tid >> 5;
    const int lm   = l & 15;
    const bool lo16 = (l < 16);
    const int h    = blockIdx.y;           // q head 0..31
    const int b    = blockIdx.z;
    const int kvh  = h >> 2;               // GROUP = 4
    const int qr0  = blockIdx.x * 128 + wid * 16;

    // ---- resident Q fragments (two K=32 chunks of head_dim 64) ----
    v16h aq[2];
    {
        const _Float16* qp =
            Qb + ((size_t)(b * GQA_S + qr0 + lm)) * GQA_DIM + h * GQA_HD;
        const int ko = lo16 ? 0 : 8;
#pragma unroll
        for (int c2 = 0; c2 < 2; ++c2)
            aq[c2] = cat16(*(const v8h*)(qp + c2 * 32 + ko),
                           *(const v8h*)(qp + c2 * 32 + 16 + ko));
    }

    v8f  o[4] = {};
    float mrow[8], lrow[8];
#pragma unroll
    for (int r = 0; r < 8; ++r) { mrow[r] = -1e30f; lrow[r] = 0.f; }

    const int nCh = (blockIdx.x + 1) * 2;  // 64-key chunks, causal bound
    const int koB = lo16 ? 0 : 8;

    v8h kReg[2], vReg[2];
    auto loadKV = [&](int ch) {
#pragma unroll
        for (int it = 0; it < 2; ++it) {
            int qq  = tid + it * 256;          // 512 half-octets per tile
            int key = qq >> 3, d8 = (qq & 7) * 8;
            const size_t base =
                ((size_t)(b * GQA_S + ch * 64 + key)) * GQA_KVD + kvh * GQA_HD + d8;
            kReg[it] = *(const v8h*)(Kb + base);
            vReg[it] = *(const v8h*)(Vb + base);
        }
    };

    loadKV(0);
    for (int ch = 0; ch < nCh; ++ch) {
        const int j0 = ch * 64;
        __syncthreads();
        {   // stage K row-major, V transposed (B layouts for both WMMAs)
#pragma unroll
            for (int it = 0; it < 2; ++it) {
                int qq  = tid + it * 256;
                int key = qq >> 3, d8 = (qq & 7) * 8;
                *(v8h*)&Ks[key][d8] = kReg[it];
#pragma unroll
                for (int j = 0; j < 8; ++j) Vt[d8 + j][key] = vReg[it][j];
            }
        }
        __syncthreads();
        if (ch + 1 < nCh) loadKV(ch + 1);  // pipeline next chunk

        if (j0 > qr0 + 15) continue;       // fully masked for this wave (uniform)

        // ---- scores: S = Q . K^T (four 16-key n-tiles, two d chunks) ----
        v8f s[4] = {};
#pragma unroll
        for (int jt = 0; jt < 4; ++jt)
#pragma unroll
            for (int c2 = 0; c2 < 2; ++c2) {
                v16h bk = cat16(*(const v8h*)&Ks[jt * 16 + lm][c2 * 32 + koB],
                                *(const v8h*)&Ks[jt * 16 + lm][c2 * 32 + koB + 16]);
                s[jt] = wmma_f16(aq[c2], bk, s[jt]);
            }

        // ---- scale (1/sqrt(64)), -2 shift, causal mask, online softmax ----
        float p[4][8], alpha[8];
#pragma unroll
        for (int r = 0; r < 8; ++r) {
            const int rowg = qr0 + (lo16 ? r : 8 + r);
            float sv[4];
#pragma unroll
            for (int jt = 0; jt < 4; ++jt) {
                sv[jt] = s[jt][r] * 0.125f - 2.0f;
                if (j0 + jt * 16 + lm > rowg) sv[jt] = -1e30f;
            }
            float mx = fmaxf(fmaxf(sv[0], sv[1]), fmaxf(sv[2], sv[3]));
#pragma unroll
            for (int d = 1; d < 16; d <<= 1)
                mx = fmaxf(mx, __shfl_xor(mx, d, 32));
            const float mn = fmaxf(mrow[r], mx);
            alpha[r] = __expf(mrow[r] - mn);
            mrow[r]  = mn;
            float ps = 0.f;
#pragma unroll
            for (int jt = 0; jt < 4; ++jt) {
                p[jt][r] = __expf(sv[jt] - mn);
                ps += p[jt][r];
            }
#pragma unroll
            for (int d = 1; d < 16; d <<= 1) ps += __shfl_xor(ps, d, 32);
            lrow[r] = lrow[r] * alpha[r] + ps;
        }
#pragma unroll
        for (int nb = 0; nb < 4; ++nb)
#pragma unroll
            for (int r = 0; r < 8; ++r) o[nb][r] *= alpha[r];

        // ---- P: D layout -> A layout via per-wave LDS staging ----
#pragma unroll
        for (int r = 0; r < 8; ++r) {
            const int rr = lo16 ? r : 8 + r;
#pragma unroll
            for (int jt = 0; jt < 4; ++jt)
                Ps[wid][rr][jt * 16 + lm] = (_Float16)p[jt][r];
        }
        const int koA = lo16 ? 0 : 8;
        v16h ap[2];
#pragma unroll
        for (int c = 0; c < 2; ++c)
            ap[c] = cat16(*(const v8h*)&Ps[wid][lm][c * 32 + koA],
                          *(const v8h*)&Ps[wid][lm][c * 32 + 16 + koA]);

        // ---- O += P x V (two 32-key chunks x four 16-wide d blocks) ----
#pragma unroll
        for (int c = 0; c < 2; ++c)
#pragma unroll
            for (int nb = 0; nb < 4; ++nb) {
                v16h bv = cat16(*(const v8h*)&Vt[nb * 16 + lm][c * 32 + koB],
                                *(const v8h*)&Vt[nb * 16 + lm][c * 32 + koB + 16]);
                o[nb] = wmma_f16(ap[c], bv, o[nb]);
            }
    }

    // ---- normalize and write attention output (f16, head-major) ----
#pragma unroll
    for (int r = 0; r < 8; ++r) {
        const int row = qr0 + (lo16 ? r : 8 + r);
        const float inv = 1.0f / lrow[r];
        _Float16* op =
            Ob + ((size_t)(b * GQA_S + row)) * GQA_DIM + h * GQA_HD + lm;
#pragma unroll
        for (int nb = 0; nb < 4; ++nb)
            op[nb * 16] = (_Float16)(o[nb][r] * inv);
    }
}

// ---------------------------------------------------------------------------
// Launch
// ---------------------------------------------------------------------------
extern "C" void kernel_launch(void* const* d_in, const int* in_sizes, int n_in,
                              void* d_out, int out_size, void* d_ws, size_t ws_size,
                              hipStream_t stream) {
    const float* q  = (const float*)d_in[0];
    const float* k  = (const float*)d_in[1];
    const float* v  = (const float*)d_in[2];
    // d_in[3] = causal mask (tril) -- applied analytically in the kernel
    const float* Wq = (const float*)d_in[4];
    const float* bq = (const float*)d_in[5];
    const float* Wk = (const float*)d_in[6];
    const float* bk = (const float*)d_in[7];
    const float* Wv = (const float*)d_in[8];
    const float* bv = (const float*)d_in[9];
    const float* Wo = (const float*)d_in[10];
    const float* bo = (const float*)d_in[11];
    float* out = (float*)d_out;

    // workspace: Q16 | K16 | V16 | attn16  (f16)
    _Float16* Q16 = (_Float16*)d_ws;
    _Float16* K16 = Q16 + (size_t)GQA_M * GQA_DIM;
    _Float16* V16 = K16 + (size_t)GQA_M * GQA_KVD;
    _Float16* A16 = V16 + (size_t)GQA_M * GQA_KVD;

    // Projections (f32 in, f16 out)
    gqa_gemm_wmma<float, _Float16><<<dim3(GQA_DIM / 128, GQA_M / 128), 256, 0, stream>>>(
        q, Wq, bq, Q16, GQA_M, GQA_DIM, GQA_DIM);
    gqa_gemm_wmma<float, _Float16><<<dim3(GQA_KVD / 128, GQA_M / 128), 256, 0, stream>>>(
        k, Wk, bk, K16, GQA_M, GQA_KVD, GQA_DIM);
    gqa_gemm_wmma<float, _Float16><<<dim3(GQA_KVD / 128, GQA_M / 128), 256, 0, stream>>>(
        v, Wv, bv, V16, GQA_M, GQA_KVD, GQA_DIM);

    // Flash attention
    gqa_flash_attn<<<dim3(GQA_S / 128, GQA_HQ, GQA_B), 256, 0, stream>>>(
        Q16, K16, V16, A16);

    // Output projection (f16 in, f32 out)
    gqa_gemm_wmma<_Float16, float><<<dim3(GQA_DIM / 128, GQA_M / 128), 256, 0, stream>>>(
        A16, Wo, bo, out, GQA_M, GQA_DIM, GQA_DIM);
}
